// Relation_5841155522972
// MI455X (gfx1250) — matrix operand order repaired and checked by reading, not compiled, not run on hardware
//
#include <hip/hip_runtime.h>

// ---------------------------------------------------------------------------
// MI455X (gfx1250, wave32) implementation of "talking-heads hard attention".
// Forward output == hard argmax gather, so we never compute attn = w @ v.
// Pipeline:
//   1) f32 -> bf16 converts (x, Wq, Wkv)
//   2) bf16 WMMA GEMMs: Q (bf16), K (bf16), V (f32); 16x128 strip per wave,
//      all 8 B-fragments loaded up front so ~16 b128 loads stay in flight
//      against the WMMA pipe (progressive s_wait_loadcnt).
//   3) stats pass: per-(b,h,i) softmax max & denom via v_wmma_f32_16x16x32_bf16
//   4) final pass: recompute scores, P tiles -> LDS, 16x16 head-mix via
//      v_wmma_f32_16x16x4_f32, LayerNorm over heads (xor-16 shuffle), argmax
//   5) gather V rows into output
// global_prefetch_b8 hints (branch-free, clamped) keep HBM/L2 ahead of compute.
// ---------------------------------------------------------------------------

typedef __attribute__((ext_vector_type(16))) __bf16 v16bf;
typedef __attribute__((ext_vector_type(8)))  float  v8f;
typedef __attribute__((ext_vector_type(2)))  float  v2f;

union Frag16 { v16bf v; uint4 q[2]; };

#define WMMA_BF16(A, B, C) \
  __builtin_amdgcn_wmma_f32_16x16x32_bf16(false, (A), false, (B), (short)0, (C), false, false)

constexpr int BATCH = 4;
constexpr int NSEQ  = 1024;
constexpr int DDIM  = 1024;
constexpr int HNUM  = 16;
constexpr int DH    = 64;
constexpr float SCALE = 0.03125f;   // DDIM^-0.5
constexpr float LN_EPS = 1e-5f;

// ---------------------------------------------------------------------------
__global__ void cvt_bf16_kernel(const float* __restrict__ in,
                                __bf16* __restrict__ out, int n) {
  int t = blockIdx.x * blockDim.x + threadIdx.x;
  if (t < n) out[t] = (__bf16)in[t];
}

// ---------------------------------------------------------------------------
// C[m][n] = sum_k A[m][k] * W[n][k]   (A: Mx1024 bf16, W rows: 1024 bf16)
// One wave computes a 16x128 strip (8 accumulators -> 8 WMMA per A-frag).
// Block = 8 waves covering all 1024 output cols; grid = M/16 blocks.
__global__ void gemm_bf16_kernel(const __bf16* __restrict__ A,
                                 const __bf16* __restrict__ W,
                                 __bf16* __restrict__ Cb,
                                 float*  __restrict__ Cf) {
  const int wave = threadIdx.x >> 5;
  const int lane = threadIdx.x & 31;
  const int lrow = lane & 15;
  const int hi   = lane >> 4;
  const int m0 = blockIdx.x * 16;
  const int n0 = wave * 128;

  const __bf16* arow = A + (size_t)(m0 + lrow) * DDIM;
  v8f acc[8] = {};
  for (int k = 0; k < DDIM; k += 32) {
    // branch-free speculative prefetch four k-steps ahead (clamped in-bounds)
    const int kpf = (k + 128 < DDIM) ? k + 128 : k;
    __builtin_prefetch(arow + kpf, 0, 3);                 // global_prefetch_b8
    __builtin_prefetch(W + (size_t)(n0 + lrow) * DDIM + kpf, 0, 3);

    Frag16 af;
    af.q[0] = *(const uint4*)(arow + k + hi * 16);
    af.q[1] = *(const uint4*)(arow + k + hi * 16 + 8);

    // issue ALL B loads first -> up to 16 b128 loads in flight, then WMMAs
    Frag16 bf[8];
#pragma unroll
    for (int nn = 0; nn < 8; ++nn) {
      const __bf16* wrow = W + (size_t)(n0 + nn * 16 + lrow) * DDIM + k + hi * 16;
      bf[nn].q[0] = *(const uint4*)(wrow);
      bf[nn].q[1] = *(const uint4*)(wrow + 8);
    }
#pragma unroll
    for (int nn = 0; nn < 8; ++nn)
      acc[nn] = WMMA_BF16(af.v, bf[nn].v, acc[nn]);
  }
#pragma unroll
  for (int nn = 0; nn < 8; ++nn) {
#pragma unroll
    for (int r = 0; r < 8; ++r) {
      const int row = m0 + r + hi * 8;
      const int col = n0 + nn * 16 + lrow;
      if (Cb) Cb[(size_t)row * DDIM + col] = (__bf16)acc[nn][r];
      else    Cf[(size_t)row * DDIM + col] = acc[nn][r];
    }
  }
}

// ---------------------------------------------------------------------------
// Pass 1: softmax row max + denominator per (b, h, i).
// Block = 8 waves, one (b, i-tile of 16) per block; wave handles heads {w, w+8}.
__global__ void attn_stats_kernel(const __bf16* __restrict__ Qb,
                                  const __bf16* __restrict__ Kb,
                                  float* __restrict__ m_ws,
                                  float* __restrict__ l_ws) {
  const int blk  = blockIdx.x;
  const int b    = blk >> 6;
  const int i0   = (blk & 63) << 4;
  const int wave = threadIdx.x >> 5;
  const int lane = threadIdx.x & 31;
  const int lrow = lane & 15;
  const int hi   = lane >> 4;

#pragma unroll
  for (int hs = 0; hs < 2; ++hs) {
    const int h = wave + hs * 8;
    const __bf16* qrow = Qb + (size_t)(b * NSEQ + i0 + lrow) * DDIM + h * DH;
    Frag16 a0, a1;
    a0.q[0] = *(const uint4*)(qrow + hi * 16);
    a0.q[1] = *(const uint4*)(qrow + hi * 16 + 8);
    a1.q[0] = *(const uint4*)(qrow + 32 + hi * 16);
    a1.q[1] = *(const uint4*)(qrow + 32 + hi * 16 + 8);

    float m_r[8], s_r[8];
#pragma unroll
    for (int r = 0; r < 8; ++r) { m_r[r] = -3.0e38f; s_r[r] = 0.f; }

    for (int jt = 0; jt < NSEQ / 16; ++jt) {
      const __bf16* krow =
          Kb + (size_t)(b * NSEQ + jt * 16 + lrow) * DDIM + h * DH + hi * 16;
      const size_t pf = (jt + 1 < NSEQ / 16) ? (size_t)16 * DDIM : 0;
      __builtin_prefetch(krow + pf, 0, 3);
      Frag16 b0, b1;
      b0.q[0] = *(const uint4*)(krow);
      b0.q[1] = *(const uint4*)(krow + 8);
      b1.q[0] = *(const uint4*)(krow + 32);
      b1.q[1] = *(const uint4*)(krow + 40);
      v8f acc = {};
      acc = WMMA_BF16(a0.v, b0.v, acc);
      acc = WMMA_BF16(a1.v, b1.v, acc);
#pragma unroll
      for (int r = 0; r < 8; ++r) {
        const float v = acc[r] * SCALE;
        const float M = fmaxf(m_r[r], v);
        s_r[r] = s_r[r] * __expf(m_r[r] - M) + __expf(v - M);
        m_r[r] = M;
      }
    }
    // reduce (max, denom) over the 16 lanes holding the same rows
#pragma unroll
    for (int mask = 1; mask <= 8; mask <<= 1) {
#pragma unroll
      for (int r = 0; r < 8; ++r) {
        const float mo = __shfl_xor(m_r[r], mask, 32);
        const float so = __shfl_xor(s_r[r], mask, 32);
        const float M = fmaxf(m_r[r], mo);
        s_r[r] = s_r[r] * __expf(m_r[r] - M) + so * __expf(mo - M);
        m_r[r] = M;
      }
    }
    if (lrow == 0) {
#pragma unroll
      for (int r = 0; r < 8; ++r) {
        const int i = i0 + r + hi * 8;
        m_ws[(b * HNUM + h) * NSEQ + i] = m_r[r];
        l_ws[(b * HNUM + h) * NSEQ + i] = s_r[r];
      }
    }
  }
}

// ---------------------------------------------------------------------------
// Pass 2: probabilities -> talking-heads mix -> LayerNorm -> running argmax.
__global__ void attn_final_kernel(const __bf16* __restrict__ Qb,
                                  const __bf16* __restrict__ Kb,
                                  const float* __restrict__ m_ws,
                                  const float* __restrict__ l_ws,
                                  const float* __restrict__ Wt,
                                  const float* __restrict__ ln_g,
                                  const float* __restrict__ ln_b,
                                  int* __restrict__ idx_ws) {
  __shared__ float P[16 * 272];   // [h][i*17 + j], padded to kill bank conflicts
  const int blk  = blockIdx.x;
  const int b    = blk >> 6;
  const int i0   = (blk & 63) << 4;
  const int wave = threadIdx.x >> 5;
  const int lane = threadIdx.x & 31;
  const int lrow = lane & 15;
  const int hi   = lane >> 4;

  // Q fragments + softmax stats for this wave's two heads
  Frag16 af[2][2];
  float mrow[2][8], linv[2][8];
#pragma unroll
  for (int hs = 0; hs < 2; ++hs) {
    const int h = wave + hs * 8;
    const __bf16* qrow = Qb + (size_t)(b * NSEQ + i0 + lrow) * DDIM + h * DH;
    af[hs][0].q[0] = *(const uint4*)(qrow + hi * 16);
    af[hs][0].q[1] = *(const uint4*)(qrow + hi * 16 + 8);
    af[hs][1].q[0] = *(const uint4*)(qrow + 32 + hi * 16);
    af[hs][1].q[1] = *(const uint4*)(qrow + 32 + hi * 16 + 8);
#pragma unroll
    for (int r = 0; r < 8; ++r) {
      const int i = i0 + r + hi * 8;
      mrow[hs][r] = m_ws[(b * HNUM + h) * NSEQ + i];
      linv[hs][r] = 1.0f / l_ws[(b * HNUM + h) * NSEQ + i];
    }
  }

#if __has_builtin(__builtin_amdgcn_wmma_f32_16x16x4_f32)
  // Wt A-fragments for f32 WMMA mix: A[m=h_out][k=h_in] = Wt[h_out][h_in]
  v2f wtf[4];
#pragma unroll
  for (int kk = 0; kk < 4; ++kk) {
    const int k0 = kk * 4 + hi * 2;
    wtf[kk][0] = Wt[lrow * 16 + k0];
    wtf[kk][1] = Wt[lrow * 16 + k0 + 1];
  }
#endif
  float g_r[8], be_r[8];
#pragma unroll
  for (int r = 0; r < 8; ++r) {
    g_r[r]  = ln_g[r + hi * 8];
    be_r[r] = ln_b[r + hi * 8];
  }

  float best[2][8];
  int   bidx[2][8];
#pragma unroll
  for (int s = 0; s < 2; ++s)
#pragma unroll
    for (int r = 0; r < 8; ++r) { best[s][r] = -3.0e38f; bidx[s][r] = 0; }

  for (int jt = 0; jt < NSEQ / 16; ++jt) {
    // scores -> probabilities for both heads of this wave
#pragma unroll
    for (int hs = 0; hs < 2; ++hs) {
      const int h = wave + hs * 8;
      const __bf16* krow =
          Kb + (size_t)(b * NSEQ + jt * 16 + lrow) * DDIM + h * DH + hi * 16;
      const size_t pf = (jt + 1 < NSEQ / 16) ? (size_t)16 * DDIM : 0;
      __builtin_prefetch(krow + pf, 0, 3);
      Frag16 b0, b1;
      b0.q[0] = *(const uint4*)(krow);
      b0.q[1] = *(const uint4*)(krow + 8);
      b1.q[0] = *(const uint4*)(krow + 32);
      b1.q[1] = *(const uint4*)(krow + 40);
      v8f acc = {};
      acc = WMMA_BF16(af[hs][0].v, b0.v, acc);
      acc = WMMA_BF16(af[hs][1].v, b1.v, acc);
#pragma unroll
      for (int r = 0; r < 8; ++r) {
        const float p = __expf(acc[r] * SCALE - mrow[hs][r]) * linv[hs][r];
        P[h * 272 + (r + hi * 8) * 17 + lrow] = p;
      }
    }
    __syncthreads();

    // head mix (16x16x16 GEMM) per i-subtile tt; each wave owns 2 tiles
#pragma unroll
    for (int slot = 0; slot < 2; ++slot) {
      const int tt = wave * 2 + slot;
      float dreg[8];
#if __has_builtin(__builtin_amdgcn_wmma_f32_16x16x4_f32)
      v8f d = {};
#pragma unroll
      for (int kk = 0; kk < 4; ++kk) {
        const int k0 = kk * 4 + hi * 2;
        v2f b2;
        b2[0] = P[(k0 + 0) * 272 + tt * 17 + lrow];
        b2[1] = P[(k0 + 1) * 272 + tt * 17 + lrow];
        d = __builtin_amdgcn_wmma_f32_16x16x4_f32(false, wtf[kk], false, b2,
                                                  (short)0, d, false, false);
      }
#pragma unroll
      for (int r = 0; r < 8; ++r) dreg[r] = d[r];
#else
#pragma unroll
      for (int r = 0; r < 8; ++r) {
        float s = 0.f;
#pragma unroll
        for (int k = 0; k < 16; ++k)
          s += Wt[(r + hi * 8) * 16 + k] * P[k * 272 + tt * 17 + lrow];
        dreg[r] = s;
      }
#endif
      // LayerNorm over the 16 heads of this (i=tt, j=lrow) column:
      // rows split across lane pairs (n, n+16) -> xor-16 shuffle combines.
      float sum = 0.f, ssq = 0.f;
#pragma unroll
      for (int r = 0; r < 8; ++r) { sum += dreg[r]; ssq += dreg[r] * dreg[r]; }
      sum += __shfl_xor(sum, 16, 32);
      ssq += __shfl_xor(ssq, 16, 32);
      const float mu = sum * 0.0625f;
      const float var = ssq * 0.0625f - mu * mu;
      const float rstd = rsqrtf(var + LN_EPS);
      const int j = jt * 16 + lrow;
#pragma unroll
      for (int r = 0; r < 8; ++r) {
        const float w2 = (dreg[r] - mu) * rstd * g_r[r] + be_r[r];
        const bool gt = w2 > best[slot][r];
        best[slot][r] = gt ? w2 : best[slot][r];
        bidx[slot][r] = gt ? j : bidx[slot][r];
      }
    }
    __syncthreads();
  }

  // argmax reduce across the 16 lanes covering different j residues
#pragma unroll
  for (int slot = 0; slot < 2; ++slot) {
#pragma unroll
    for (int mask = 1; mask <= 8; mask <<= 1) {
#pragma unroll
      for (int r = 0; r < 8; ++r) {
        const float vo = __shfl_xor(best[slot][r], mask, 32);
        const int   jo = __shfl_xor(bidx[slot][r], mask, 32);
        const bool take = (vo > best[slot][r]) ||
                          (vo == best[slot][r] && jo < bidx[slot][r]);
        best[slot][r] = take ? vo : best[slot][r];
        bidx[slot][r] = take ? jo : bidx[slot][r];
      }
    }
    if (lrow == 0) {
      const int i = i0 + wave * 2 + slot;
#pragma unroll
      for (int r = 0; r < 8; ++r)
        idx_ws[(b * HNUM + (r + hi * 8)) * NSEQ + i] = bidx[slot][r];
    }
  }
}

// ---------------------------------------------------------------------------
// out[b, i, h*64 + d] = V[b*N + idx(b,h,i)][h*64 + d]   (float4 vectorized)
__global__ void gather_out_kernel(const float* __restrict__ Vf,
                                  const int* __restrict__ idx_ws,
                                  float* __restrict__ out) {
  const int t = blockIdx.x * blockDim.x + threadIdx.x;   // 1M threads
  const int col = (t * 4) & (DDIM - 1);
  const int row = (t * 4) >> 10;            // b*N + i
  const int b = row >> 10;
  const int i = row & (NSEQ - 1);
  const int h = col >> 6;
  const int idx = idx_ws[(b * HNUM + h) * NSEQ + i];
  const float4 v = *(const float4*)(Vf + (size_t)(b * NSEQ + idx) * DDIM + col);
  *(float4*)(out + (size_t)t * 4) = v;
}

// ---------------------------------------------------------------------------
extern "C" void kernel_launch(void* const* d_in, const int* in_sizes, int n_in,
                              void* d_out, int out_size, void* d_ws, size_t ws_size,
                              hipStream_t stream) {
  const float* x    = (const float*)d_in[0];
  const float* Wq   = (const float*)d_in[1];
  const float* Wkv  = (const float*)d_in[2];
  const float* Wt   = (const float*)d_in[3];
  const float* ln_g = (const float*)d_in[4];
  const float* ln_b = (const float*)d_in[5];
  float* out = (float*)d_out;

  char* ws = (char*)d_ws;
  size_t off = 0;
  __bf16* xb   = (__bf16*)(ws + off); off += (size_t)4096 * 1024 * 2;  // 8 MB
  __bf16* wqb  = (__bf16*)(ws + off); off += (size_t)1024 * 1024 * 2;  // 2 MB
  __bf16* wkvb = (__bf16*)(ws + off); off += (size_t)2048 * 1024 * 2;  // 4 MB
  __bf16* Qb   = (__bf16*)(ws + off); off += (size_t)4096 * 1024 * 2;  // 8 MB
  __bf16* Kb   = (__bf16*)(ws + off); off += (size_t)4096 * 1024 * 2;  // 8 MB
  float*  Vf   = (float*)(ws + off);  off += (size_t)4096 * 1024 * 4;  // 16 MB
  float*  m_ws = (float*)(ws + off);  off += (size_t)65536 * 4;
  float*  l_ws = (float*)(ws + off);  off += (size_t)65536 * 4;
  int*    idx_ws = (int*)(ws + off);  off += (size_t)65536 * 4;
  (void)ws_size; (void)in_sizes; (void)n_in; (void)out_size;

  cvt_bf16_kernel<<<16384, 256, 0, stream>>>(x, xb, 4096 * 1024);
  cvt_bf16_kernel<<<4096, 256, 0, stream>>>(Wq, wqb, 1024 * 1024);
  cvt_bf16_kernel<<<8192, 256, 0, stream>>>(Wkv, wkvb, 2048 * 1024);

  gemm_bf16_kernel<<<256, 256, 0, stream>>>(xb, wqb, Qb, nullptr);
  gemm_bf16_kernel<<<256, 256, 0, stream>>>(xb, wkvb, Kb, nullptr);
  gemm_bf16_kernel<<<256, 256, 0, stream>>>(
      xb, wkvb + (size_t)1024 * 1024, nullptr, Vf);

  attn_stats_kernel<<<256, 256, 0, stream>>>(Qb, Kb, m_ws, l_ws);
  attn_final_kernel<<<256, 256, 0, stream>>>(Qb, Kb, m_ws, l_ws, Wt, ln_g,
                                             ln_b, idx_ws);
  gather_out_kernel<<<4096, 256, 0, stream>>>(Vf, idx_ws, out);
}
